// MinkowskiSparseBottleneckV2_26972394619246
// MI455X (gfx1250) — matrix operand order repaired and checked by reading, not compiled, not run on hardware
//
#include <hip/hip_runtime.h>
#include <hip/hip_bf16.h>

// ---------------------------------------------------------------------------
// MinkowskiSparseBottleneckV2 on gfx1250 (MI455X, wave32, WMMA).
//
// Pipeline (all GEMM math on v_wmma_f32_16x16x32_bf16, f32 accumulate):
//   K0: fold BN1/2/3 into scale/shift; cast+transpose w1,w2,w3 to bf16 so the
//       WMMA B-operand is a contiguous 32B per-lane load (Bt[n][k] layout).
//   K1: h2 = bf16(relu(bn2( relu(bn1(x)) @ w1 + b1 )))          [N,64] bf16
//   K2: h3 = bf16(relu(bn3( sum_k gather(h2)*mask @ w2[k] + b2 )))
//       (h2 is 33MB -> the random gather hits the 192MB L2, not HBM)
//   K3: out = h3 @ w3 + b3 + relu(bn1(x))   (shortcut recomputed in f32)
//
// WMMA operand layouts used here (wave32; A and B share contiguous k order):
//   A (bf16 16x32): lane L holds row M=L%16; K runs [kb0,kb0+8) and
//                   [kb0+8,kb0+16), kb0 = (L>=16 ? 16 : 0)  -> two 16B loads.
//   B (bf16 32x16): lane L holds col N=L%16; element e -> K=e+(L>=16?16:0)
//                   -> one contiguous 32B load from transposed weights.
//   C/D (f32 16x16): lane L holds col N=L%16; VGPR r -> row M=r+(L>=16?8:0).
// ---------------------------------------------------------------------------

typedef __attribute__((ext_vector_type(16))) __bf16 v16bf;
typedef __attribute__((ext_vector_type(8)))  __bf16 bf16x8;
typedef __attribute__((ext_vector_type(8)))  float  v8f;
typedef __attribute__((ext_vector_type(4)))  float  f32x4;

#define NROWS 262144
#define CIN   256
#define MID   64
#define KOFF  9

// ---- workspace layout (bytes); total ~67.3 MB ----
#define H2_OFF   0u                       // N*64 bf16  = 33554432
#define H3_OFF   33554432u                // N*64 bf16  = 33554432
#define W1T_OFF  67108864u                // 64*256 bf16
#define W2T_OFF  67141632u                // 9*64*64 bf16
#define W3T_OFF  67215360u                // 256*64 bf16
#define SC1_OFF  67248128u                // 256 f32
#define SH1_OFF  67249152u
#define SC2_OFF  67250176u                // 64 f32 (padded)
#define SH2_OFF  67251200u
#define SC3_OFF  67252224u
#define SH3_OFF  67253248u

static __device__ inline v16bf zero_frag() {
  v16bf z;
#pragma unroll
  for (int i = 0; i < 16; ++i) z[i] = (__bf16)0.0f;
  return z;
}

// A-fragment from a bf16 row: two contiguous 8-element runs.
static __device__ inline v16bf load_a_bf16(const __bf16* rowp, int kk) {
  bf16x8 lo = *(const bf16x8*)(rowp + kk);
  bf16x8 hi = *(const bf16x8*)(rowp + kk + 8);
  v16bf a;
#pragma unroll
  for (int i = 0; i < 8; ++i) { a[i] = lo[i]; a[i + 8] = hi[i]; }
  return a;
}

// A-fragment from f32 x row with fused bn1+relu, converted to bf16.
static __device__ inline v16bf bn_relu_frag(const float* __restrict__ xrow,
                                            const float* __restrict__ sc,
                                            const float* __restrict__ sh,
                                            int kk) {
  v16bf a;
#pragma unroll
  for (int h = 0; h < 2; ++h) {
    int o = kk + h * 8;
    f32x4 x0 = *(const f32x4*)(xrow + o);
    f32x4 x1 = *(const f32x4*)(xrow + o + 4);
    f32x4 s0 = *(const f32x4*)(sc + o);
    f32x4 s1 = *(const f32x4*)(sc + o + 4);
    f32x4 t0 = *(const f32x4*)(sh + o);
    f32x4 t1 = *(const f32x4*)(sh + o + 4);
#pragma unroll
    for (int i = 0; i < 4; ++i) {
      a[h * 8 + i]     = (__bf16)fmaxf(x0[i] * s0[i] + t0[i], 0.0f);
      a[h * 8 + 4 + i] = (__bf16)fmaxf(x1[i] * s1[i] + t1[i], 0.0f);
    }
  }
  return a;
}

// ---------------------------------------------------------------------------
// K0: BN folding + weight cast/transpose. 70016 work items.
// ---------------------------------------------------------------------------
__global__ void k0_prep(const float* __restrict__ g1, const float* __restrict__ be1,
                        const float* __restrict__ mu1, const float* __restrict__ va1,
                        const float* __restrict__ g2, const float* __restrict__ be2,
                        const float* __restrict__ mu2, const float* __restrict__ va2,
                        const float* __restrict__ g3, const float* __restrict__ be3,
                        const float* __restrict__ mu3, const float* __restrict__ va3,
                        const float* __restrict__ w1, const float* __restrict__ w2,
                        const float* __restrict__ w3,
                        float* __restrict__ sc1, float* __restrict__ sh1,
                        float* __restrict__ sc2, float* __restrict__ sh2,
                        float* __restrict__ sc3, float* __restrict__ sh3,
                        __bf16* __restrict__ w1t, __bf16* __restrict__ w2t,
                        __bf16* __restrict__ w3t) {
  int t = blockIdx.x * 256 + threadIdx.x;
  if (t < 256) {  // bn1
    float s = g1[t] * rsqrtf(va1[t] + 1e-5f);
    sc1[t] = s; sh1[t] = be1[t] - mu1[t] * s; return;
  }
  t -= 256;
  if (t < 64) {   // bn2
    float s = g2[t] * rsqrtf(va2[t] + 1e-5f);
    sc2[t] = s; sh2[t] = be2[t] - mu2[t] * s; return;
  }
  t -= 64;
  if (t < 64) {   // bn3
    float s = g3[t] * rsqrtf(va3[t] + 1e-5f);
    sc3[t] = s; sh3[t] = be3[t] - mu3[t] * s; return;
  }
  t -= 64;
  if (t < 64 * 256) {  // w1t[m][c] = w1[c][m]
    int m = t >> 8, c = t & 255;
    w1t[t] = (__bf16)w1[c * 64 + m]; return;
  }
  t -= 64 * 256;
  if (t < 9 * 64 * 64) {  // w2t[k][d][c] = w2[k][c][d]
    int k = t >> 12, r = t & 4095, d = r >> 6, c = r & 63;
    w2t[t] = (__bf16)w2[k * 4096 + c * 64 + d]; return;
  }
  t -= 9 * 64 * 64;
  if (t < 256 * 64) {  // w3t[n][c] = w3[c][n]
    int n = t >> 6, c = t & 63;
    w3t[t] = (__bf16)w3[c * 256 + n]; return;
  }
}

// ---------------------------------------------------------------------------
// K1: h2 = bf16(relu(bn2(relu(bn1(x)) @ w1 + b1))).  One wave per 16x64 tile.
// ---------------------------------------------------------------------------
__global__ __launch_bounds__(256) void k1_gemm1(
    const float* __restrict__ x, const float* __restrict__ sc1,
    const float* __restrict__ sh1, const __bf16* __restrict__ w1t,
    const float* __restrict__ b1, const float* __restrict__ sc2,
    const float* __restrict__ sh2, __bf16* __restrict__ h2) {
  const int wave = threadIdx.x >> 5;
  const int lane = threadIdx.x & 31;
  const int l16 = lane & 15, half = lane >> 4;
  const int kb0 = half * 16;
  const int rowBase = (blockIdx.x * 8 + wave) * 16;
  const float* xrow = x + (size_t)(rowBase + l16) * CIN;

  v8f acc[4] = {};
#pragma unroll
  for (int ks = 0; ks < CIN; ks += 32) {
    v16bf a = bn_relu_frag(xrow, sc1, sh1, ks + kb0);
#pragma unroll
    for (int nt = 0; nt < 4; ++nt) {
      const __bf16* bp = w1t + (nt * 16 + l16) * CIN + ks + half * 16;
      v16bf b = *(const v16bf*)bp;
      acc[nt] = __builtin_amdgcn_wmma_f32_16x16x32_bf16(
          false, a, false, b, (short)0, acc[nt], false, false);
    }
  }
#pragma unroll
  for (int nt = 0; nt < 4; ++nt) {
    int col = nt * 16 + l16;
    float bb = b1[col], s = sc2[col], t = sh2[col];
#pragma unroll
    for (int r = 0; r < 8; ++r) {
      int m = rowBase + r + half * 8;
      float v = fmaxf(s * (acc[nt][r] + bb) + t, 0.0f);
      h2[(size_t)m * MID + col] = (__bf16)v;
    }
  }
}

// ---------------------------------------------------------------------------
// K2: gather-GEMM 3x3 sparse conv + bn3 + relu -> h3 (bf16).
// nbr_mask comes from jax bernoulli (bool_, 1 byte/elem) -> read as uint8.
// ---------------------------------------------------------------------------
__global__ __launch_bounds__(256) void k2_conv(
    const __bf16* __restrict__ h2, const __bf16* __restrict__ w2t,
    const int* __restrict__ nbr_idx, const unsigned char* __restrict__ nbr_mask,
    const float* __restrict__ b2, const float* __restrict__ sc3,
    const float* __restrict__ sh3, __bf16* __restrict__ h3) {
  const int wave = threadIdx.x >> 5;
  const int lane = threadIdx.x & 31;
  const int l16 = lane & 15, half = lane >> 4;
  const int kb0 = half * 16;
  const int rowBase = (blockIdx.x * 8 + wave) * 16;
  const int row = rowBase + l16;

  v8f acc[4] = {};
  for (int k = 0; k < KOFF; ++k) {
    int src = nbr_idx[row * KOFF + k];
    bool mk = nbr_mask[row * KOFF + k] != 0;
    const __bf16* hrow = h2 + (size_t)src * MID;
#pragma unroll
    for (int ks = 0; ks < MID; ks += 32) {
      v16bf a = zero_frag();
      if (mk) a = load_a_bf16(hrow, ks + kb0);
#pragma unroll
      for (int nt = 0; nt < 4; ++nt) {
        const __bf16* bp = w2t + k * 4096 + (nt * 16 + l16) * MID + ks + half * 16;
        v16bf b = *(const v16bf*)bp;
        acc[nt] = __builtin_amdgcn_wmma_f32_16x16x32_bf16(
            false, a, false, b, (short)0, acc[nt], false, false);
      }
    }
  }
#pragma unroll
  for (int nt = 0; nt < 4; ++nt) {
    int col = nt * 16 + l16;
    float bb = b2[col], s = sc3[col], t = sh3[col];
#pragma unroll
    for (int r = 0; r < 8; ++r) {
      int m = rowBase + r + half * 8;
      float v = fmaxf(s * (acc[nt][r] + bb) + t, 0.0f);
      h3[(size_t)m * MID + col] = (__bf16)v;
    }
  }
}

// ---------------------------------------------------------------------------
// K3: out = h3 @ w3 + b3 + relu(bn1(x)).  Wave -> 16 rows x 64 cols;
// 4 waves cover the 256 output cols, 2 row-tiles per block.
// ---------------------------------------------------------------------------
__global__ __launch_bounds__(256) void k3_gemm3(
    const __bf16* __restrict__ h3, const __bf16* __restrict__ w3t,
    const float* __restrict__ b3, const float* __restrict__ x,
    const float* __restrict__ sc1, const float* __restrict__ sh1,
    float* __restrict__ out) {
  const int wave = threadIdx.x >> 5;
  const int lane = threadIdx.x & 31;
  const int l16 = lane & 15, half = lane >> 4;
  const int kb0 = half * 16;
  const int rowBase = (blockIdx.x * 2 + (wave >> 2)) * 16;
  const int colQ = wave & 3;
  const __bf16* hrow = h3 + (size_t)(rowBase + l16) * MID;

  v8f acc[4] = {};
#pragma unroll
  for (int ks = 0; ks < MID; ks += 32) {
    v16bf a = load_a_bf16(hrow, ks + kb0);
#pragma unroll
    for (int nt = 0; nt < 4; ++nt) {
      int n = colQ * 64 + nt * 16 + l16;
      v16bf b = *(const v16bf*)(w3t + n * MID + ks + half * 16);
      acc[nt] = __builtin_amdgcn_wmma_f32_16x16x32_bf16(
          false, a, false, b, (short)0, acc[nt], false, false);
    }
  }
#pragma unroll
  for (int nt = 0; nt < 4; ++nt) {
    int col = colQ * 64 + nt * 16 + l16;
    float bb = b3[col], s = sc1[col], t = sh1[col];
#pragma unroll
    for (int r = 0; r < 8; ++r) {
      int m = rowBase + r + half * 8;
      float shortcut = fmaxf(x[(size_t)m * CIN + col] * s + t, 0.0f);
      out[(size_t)m * CIN + col] = acc[nt][r] + bb + shortcut;
    }
  }
}

// ---------------------------------------------------------------------------
extern "C" void kernel_launch(void* const* d_in, const int* in_sizes, int n_in,
                              void* d_out, int out_size, void* d_ws, size_t ws_size,
                              hipStream_t stream) {
  const float* x   = (const float*)d_in[0];
  const float* g1  = (const float*)d_in[1];
  const float* be1 = (const float*)d_in[2];
  const float* mu1 = (const float*)d_in[3];
  const float* va1 = (const float*)d_in[4];
  const float* w1  = (const float*)d_in[5];
  const float* b1  = (const float*)d_in[6];
  const float* g2  = (const float*)d_in[7];
  const float* be2 = (const float*)d_in[8];
  const float* mu2 = (const float*)d_in[9];
  const float* va2 = (const float*)d_in[10];
  const float* w2  = (const float*)d_in[11];
  const float* b2  = (const float*)d_in[12];
  const float* g3  = (const float*)d_in[13];
  const float* be3 = (const float*)d_in[14];
  const float* mu3 = (const float*)d_in[15];
  const float* va3 = (const float*)d_in[16];
  const float* w3  = (const float*)d_in[17];
  const float* b3  = (const float*)d_in[18];
  const int*   nbr_idx  = (const int*)d_in[19];
  const unsigned char* nbr_mask = (const unsigned char*)d_in[20];
  float* out = (float*)d_out;

  char* ws = (char*)d_ws;
  __bf16* h2  = (__bf16*)(ws + H2_OFF);
  __bf16* h3  = (__bf16*)(ws + H3_OFF);
  __bf16* w1t = (__bf16*)(ws + W1T_OFF);
  __bf16* w2t = (__bf16*)(ws + W2T_OFF);
  __bf16* w3t = (__bf16*)(ws + W3T_OFF);
  float* sc1 = (float*)(ws + SC1_OFF);
  float* sh1 = (float*)(ws + SH1_OFF);
  float* sc2 = (float*)(ws + SC2_OFF);
  float* sh2 = (float*)(ws + SH2_OFF);
  float* sc3 = (float*)(ws + SC3_OFF);
  float* sh3 = (float*)(ws + SH3_OFF);

  k0_prep<<<274, 256, 0, stream>>>(g1, be1, mu1, va1, g2, be2, mu2, va2,
                                   g3, be3, mu3, va3, w1, w2, w3,
                                   sc1, sh1, sc2, sh2, sc3, sh3, w1t, w2t, w3t);
  k1_gemm1<<<NROWS / 128, 256, 0, stream>>>(x, sc1, sh1, w1t, b1, sc2, sh2, h2);
  k2_conv<<<NROWS / 128, 256, 0, stream>>>(h2, w2t, nbr_idx, nbr_mask, b2, sc3, sh3, h3);
  k3_gemm3<<<NROWS / 32, 256, 0, stream>>>(h3, w3t, b3, x, sc1, sh1, out);
}
